// HymbaAttention_39668317946148
// MI455X (gfx1250) — matrix operand order repaired and checked by reading, not compiled, not run on hardware
//
#include <hip/hip_runtime.h>
#include <math.h>

// ---------------- problem constants ----------------
#define D_MODEL 2048
#define N_HEADS 16
#define N_KV    4
#define HD      128
#define SEQ     2048
#define BATCH   2
#define BT      (BATCH*SEQ)   // 4096
#define NQKV    3072          // 2048 q + 512 k + 512 v
#define WINDOW  1024
#define NUM_META 128

// ---------------- WMMA helpers (CDNA5, wave32) ----------------
typedef __attribute__((ext_vector_type(16))) __bf16 v16bf;
typedef __attribute__((ext_vector_type(8)))  float  v8f;

union FragU { uint4 u[2]; v16bf v; };

static __device__ __forceinline__ v16bf mkfrag(uint4 lo, uint4 hi) {
  FragU f; f.u[0] = lo; f.u[1] = hi; return f.v;
}
static __device__ __forceinline__ v8f wmma_bf16(v16bf a, v16bf b, v8f c) {
  // (neg_a, A, neg_b, B, c_mod, C, reuse_a, reuse_b)
  return __builtin_amdgcn_wmma_f32_16x16x32_bf16(false, a, false, b, (short)0, c, false, false);
}
static __device__ __forceinline__ __bf16 to_bf16(float f) { return (__bf16)f; }

// ---------------- CDNA5 async global->LDS copy (ASYNCcnt path) ----------------
// Each lane copies 16B from its global address to its LDS byte offset.
// Generic pointers to LDS carry the LDS offset in the low 32 bits (ISA 10.2).
static __device__ __forceinline__ void async_load_b128(void* lds_ptr, const void* gptr) {
  unsigned lds_off = (unsigned)(uintptr_t)lds_ptr;
  asm volatile("global_load_async_to_lds_b128 %0, %1, off"
               :: "v"(lds_off), "v"(gptr) : "memory");
}
static __device__ __forceinline__ void wait_async0() {
  asm volatile("s_wait_asynccnt 0" ::: "memory");
}

// ---------------- elementwise f32 -> bf16 ----------------
__global__ void cvt_bf16_kernel(const float* __restrict__ src, __bf16* __restrict__ dst, int n) {
  int i = (blockIdx.x * 256 + threadIdx.x) * 4;
  if (i + 3 < n) {
    float4 f = *(const float4*)&src[i];
    dst[i + 0] = to_bf16(f.x);
    dst[i + 1] = to_bf16(f.y);
    dst[i + 2] = to_bf16(f.z);
    dst[i + 3] = to_bf16(f.w);
  }
}

// ---------------- W[K=2048][N] -> Wt_bf16[(rowOff+n)][k] (K-contiguous) ----------------
__global__ void transpose_cvt_kernel(const float* __restrict__ src, __bf16* __restrict__ dst,
                                     int N, int rowOff) {
  __shared__ float tile[32][33];
  int n0 = blockIdx.x * 32, k0 = blockIdx.y * 32;
  int tx = threadIdx.x, ty = threadIdx.y;
#pragma unroll
  for (int i = 0; i < 4; ++i)
    tile[ty + 8 * i][tx] = src[(size_t)(k0 + ty + 8 * i) * N + n0 + tx];
  __syncthreads();
#pragma unroll
  for (int i = 0; i < 4; ++i)
    dst[(size_t)(rowOff + n0 + ty + 8 * i) * D_MODEL + k0 + tx] = to_bf16(tile[tx][ty + 8 * i]);
}

// ---------------- bf16 WMMA GEMM: C[M][N] = A[M][K] * Bt[N][K]^T ----------------
// block = 128 threads (4 waves, 2x2), tile 128x128, each wave 64x64 (16 WMMAs/kstep),
// double-buffered LDS fed by async global->LDS copies.
__global__ __launch_bounds__(128)
void gemm_bf16_kernel(const __bf16* __restrict__ A, const __bf16* __restrict__ Bt,
                      float* __restrict__ C, int M, int N, int K) {
  __shared__ alignas(16) __bf16 As[2][128][64];
  __shared__ alignas(16) __bf16 Bs[2][128][64];
  int tid = threadIdx.x;
  int lane = tid & 31, w = tid >> 5;
  int wm = w >> 1, wn = w & 1;           // 2x2 wave grid -> 64x64 per wave
  int r = lane & 15, hh = lane >> 4;
  int m0 = blockIdx.y * 128, n0 = blockIdx.x * 128;

  // 1024 16B-chunks per tile; 128 threads -> 8 chunks each per tile
  auto load_tiles = [&](int buf, int k0) {
#pragma unroll
    for (int c = 0; c < 8; ++c) {
      int chunk = tid * 8 + c;
      int row = chunk >> 3, off = (chunk & 7) * 8;
      async_load_b128(&As[buf][row][off], &A[(size_t)(m0 + row) * K + k0 + off]);
      async_load_b128(&Bs[buf][row][off], &Bt[(size_t)(n0 + row) * K + k0 + off]);
    }
  };

  v8f acc[4][4] = {};

  load_tiles(0, 0);
  wait_async0();
  __syncthreads();

  for (int k0 = 0; k0 < K; k0 += 64) {
    int cur = (k0 >> 6) & 1;
    if (k0 + 64 < K) load_tiles(cur ^ 1, k0 + 64);  // prefetch next tile behind compute

#pragma unroll
    for (int kk = 0; kk < 64; kk += 32) {
      v16bf af[4], bf[4];
#pragma unroll
      for (int i = 0; i < 4; ++i) {
        int row = wm * 64 + i * 16 + r;
        uint4 lo = *(const uint4*)&As[cur][row][kk + hh * 16];
        uint4 hi = *(const uint4*)&As[cur][row][kk + hh * 16 + 8];
        af[i] = mkfrag(lo, hi);
      }
#pragma unroll
      for (int j = 0; j < 4; ++j) {
        int col = wn * 64 + j * 16 + r;
        const uint4* p = (const uint4*)&Bs[cur][col][kk + hh * 16];
        bf[j] = mkfrag(p[0], p[1]);
      }
#pragma unroll
      for (int i = 0; i < 4; ++i)
#pragma unroll
        for (int j = 0; j < 4; ++j)
          acc[i][j] = wmma_bf16(af[i], bf[j], acc[i][j]);
    }
    wait_async0();
    __syncthreads();
  }

  // C layout: VGPR g -> row = hh*8+g, col = lane&15
#pragma unroll
  for (int i = 0; i < 4; ++i)
#pragma unroll
    for (int j = 0; j < 4; ++j)
#pragma unroll
      for (int g = 0; g < 8; ++g) {
        int row = m0 + wm * 64 + i * 16 + hh * 8 + g;
        int col = n0 + wn * 64 + j * 16 + r;
        C[(size_t)row * N + col] = acc[i][j][g];
      }
}

// ---------------- per-head RMSNorm + RoPE, f32 QKV -> bf16 Q/K ----------------
// grid (BT, 20): y<16 -> q head, y>=16 -> kv head.  block 128 (one per dim elem)
__global__ __launch_bounds__(128)
void qk_norm_rope_kernel(const float* __restrict__ QKV, const int* __restrict__ pos_ids,
                         const float* __restrict__ qw, const float* __restrict__ kw,
                         __bf16* __restrict__ Qb, __bf16* __restrict__ Kb) {
  int row = blockIdx.x;          // 0..BT-1
  int h   = blockIdx.y;          // 0..19
  int e   = threadIdx.x;         // 0..127
  int b = row >> 11, t = row & (SEQ - 1);
  bool isQ = (h < N_HEADS);
  int col = isQ ? h * HD + e : D_MODEL + (h - N_HEADS) * HD + e;
  float x = QKV[(size_t)row * NQKV + col];

  __shared__ float sh[128];
  __shared__ float xs[128];
  sh[e] = x * x;
  __syncthreads();
  for (int s = 64; s > 0; s >>= 1) {
    if (e < s) sh[e] += sh[e + s];
    __syncthreads();
  }
  float rms = rsqrtf(sh[0] * (1.0f / (float)HD) + 1e-6f);
  float wv = isQ ? qw[e] : kw[e];
  float xn = wv * x * rms;
  xs[e] = xn;
  __syncthreads();

  int i = e >> 1;
  float pos = (float)(isQ ? pos_ids[t] : t);
  float inv = __powf(10000.0f, -(float)i * (1.0f / 64.0f));
  float fr = pos * inv;
  float sn, cs;
  __sincosf(fr, &sn, &cs);
  float x1 = xs[e & ~1], x2 = xs[e | 1];
  float outv = (e & 1) ? (x1 * sn + x2 * cs) : (x1 * cs - x2 * sn);

  if (isQ) {
    size_t o = (((size_t)(b * N_HEADS + h) * SEQ) + t) * HD + e;
    Qb[o] = to_bf16(outv);
  } else {
    size_t o = (((size_t)(b * N_KV + (h - N_HEADS)) * SEQ) + t) * HD + e;
    Kb[o] = to_bf16(outv);
  }
}

// ---------------- V: QKV f32 [row][2560+hk*128+d] -> VT bf16 [(b,hk)][d][t] ----------------
__global__ __launch_bounds__(256)
void v_transpose_kernel(const float* __restrict__ QKV, __bf16* __restrict__ VT) {
  int id = blockIdx.x;                 // 0..1023
  int drow = id & 127, bh = id >> 7;   // bh = b*4 + hk
  const float* src = QKV + (size_t)(bh >> 2) * SEQ * NQKV + (D_MODEL + 512) + (bh & 3) * HD + drow;
  __bf16* dst = VT + ((size_t)bh * HD + drow) * SEQ;
  for (int t = threadIdx.x; t < SEQ; t += 256)
    dst[t] = to_bf16(src[(size_t)t * NQKV]);
}

// ---------------- flash attention: sinks + sliding window, GQA ----------------
// grid (SEQ/64, N_HEADS, BATCH), block 128 (4 waves, 16 query rows each)
__global__ __launch_bounds__(128)
void attn_kernel(const __bf16* __restrict__ Qb, const __bf16* __restrict__ Kb,
                 const __bf16* __restrict__ VTb, __bf16* __restrict__ Ob) {
  int qb = blockIdx.x * 64;
  int hq = blockIdx.y, b = blockIdx.z;
  int hk = hq >> 2;                     // N_REP = 4
  int tid = threadIdx.x, lane = tid & 31, w = tid >> 5;
  int r = lane & 15, hh = lane >> 4;

  __shared__ alignas(16) __bf16 Ks[64][128];   // [key][d]
  __shared__ alignas(16) __bf16 Vs[128][64];   // [d][key]
  __shared__ alignas(16) __bf16 Ps[4][16][64]; // per-wave P tile

  const size_t qBase = (size_t)(b * N_HEADS + hq) * SEQ * HD;
  const size_t kBase = (size_t)(b * N_KV + hk) * SEQ * HD;
  const size_t vBase = (size_t)(b * N_KV + hk) * HD * SEQ;

  // Q fragments for this wave's 16 rows (A layout), kept resident
  v16bf qf[4];
  int qrow = qb + w * 16 + r;
#pragma unroll
  for (int kd = 0; kd < 4; ++kd) {
    const __bf16* p = &Qb[qBase + (size_t)qrow * HD + kd * 32];
    uint4 lo = *(const uint4*)&p[hh * 16];
    uint4 hi = *(const uint4*)&p[hh * 16 + 8];
    qf[kd] = mkfrag(lo, hi);
  }

  float m_i[8], l_i[8];
#pragma unroll
  for (int g = 0; g < 8; ++g) { m_i[g] = -__builtin_inff(); l_i[g] = 0.0f; }
  v8f o[8] = {};

  const int qmax = qb + 63;
  const int winStart = (qb - (WINDOW - 1) > NUM_META) ? qb - (WINDOW - 1) : NUM_META;
  const float scale = 0.08838834764831845f;  // 1/sqrt(128)

  for (int kb = 0; kb <= qmax; kb += 64) {
    bool need = (kb < NUM_META) || (kb + 63 >= winStart);
    if (!need) continue;

    // async-load K tile (64x128) and V^T tile (128x64) into LDS
    {
      int row = tid >> 1, off = (tid & 1) * 64;
      const __bf16* sk = &Kb[kBase + (size_t)(kb + row) * HD + off];
      const __bf16* sv = &VTb[vBase + (size_t)tid * SEQ + kb];
#pragma unroll
      for (int c = 0; c < 8; ++c) {
        async_load_b128(&Ks[row][off + c * 8], sk + c * 8);
        async_load_b128(&Vs[tid][c * 8], sv + c * 8);
      }
    }
    wait_async0();
    __syncthreads();

    // scores S = Q K^T  (16 x 64 per wave)
    v8f s[4] = {};
#pragma unroll
    for (int j = 0; j < 4; ++j) {
#pragma unroll
      for (int kd = 0; kd < 4; ++kd) {
        const uint4* p = (const uint4*)&Ks[j * 16 + r][kd * 32 + hh * 16];
        s[j] = wmma_bf16(qf[kd], mkfrag(p[0], p[1]), s[j]);
      }
    }

    // scale + mask
#pragma unroll
    for (int j = 0; j < 4; ++j) {
      int tk = kb + j * 16 + r;
#pragma unroll
      for (int g = 0; g < 8; ++g) {
        int tq = qb + w * 16 + hh * 8 + g;
        float val = s[j][g] * scale;
        bool ok = (tk < NUM_META) || (tk >= tq - (WINDOW - 1) && tk <= tq);
        s[j][g] = ok ? val : -__builtin_inff();
      }
    }

    // online softmax (rows live in 16-lane halves)
    float alpha[8];
#pragma unroll
    for (int g = 0; g < 8; ++g) {
      float mx = fmaxf(fmaxf(s[0][g], s[1][g]), fmaxf(s[2][g], s[3][g]));
#pragma unroll
      for (int msk = 1; msk < 16; msk <<= 1) mx = fmaxf(mx, __shfl_xor(mx, msk, 32));
      float mnew = fmaxf(m_i[g], mx);
      alpha[g] = __expf(m_i[g] - mnew);
      m_i[g] = mnew;
      float rs = 0.0f;
#pragma unroll
      for (int j = 0; j < 4; ++j) {
        float pv = __expf(s[j][g] - mnew);
        rs += pv;
        Ps[w][hh * 8 + g][j * 16 + r] = to_bf16(pv);
      }
#pragma unroll
      for (int msk = 1; msk < 16; msk <<= 1) rs += __shfl_xor(rs, msk, 32);
      l_i[g] = l_i[g] * alpha[g] + rs;
    }
#pragma unroll
    for (int jd = 0; jd < 8; ++jd)
#pragma unroll
      for (int g = 0; g < 8; ++g) o[jd][g] = o[jd][g] * alpha[g];

    // O += P V  (A frag from Ps, B frag from Vs)
#pragma unroll
    for (int kk = 0; kk < 64; kk += 32) {
      uint4 lo = *(const uint4*)&Ps[w][r][kk + hh * 16];
      uint4 hi = *(const uint4*)&Ps[w][r][kk + hh * 16 + 8];
      v16bf pf = mkfrag(lo, hi);
#pragma unroll
      for (int jd = 0; jd < 8; ++jd) {
        const uint4* p = (const uint4*)&Vs[jd * 16 + r][kk + hh * 16];
        o[jd] = wmma_bf16(pf, mkfrag(p[0], p[1]), o[jd]);
      }
    }
    __syncthreads();
  }

  // write attention output (bf16, row-major d-contiguous for the Wo GEMM)
#pragma unroll
  for (int jd = 0; jd < 8; ++jd)
#pragma unroll
    for (int g = 0; g < 8; ++g) {
      int t = qb + w * 16 + hh * 8 + g;
      float val = o[jd][g] / l_i[g];
      Ob[((size_t)(b * SEQ + t)) * D_MODEL + hq * HD + jd * 16 + r] = to_bf16(val);
    }
}

// ---------------- host side ----------------
extern "C" void kernel_launch(void* const* d_in, const int* in_sizes, int n_in,
                              void* d_out, int out_size, void* d_ws, size_t ws_size,
                              hipStream_t stream) {
  (void)in_sizes; (void)n_in; (void)out_size; (void)ws_size;
  const float* hidden = (const float*)d_in[0];
  const int*   pos    = (const int*)d_in[1];
  const float* Wq     = (const float*)d_in[2];
  const float* Wk     = (const float*)d_in[3];
  const float* Wv     = (const float*)d_in[4];
  const float* Wo     = (const float*)d_in[5];
  const float* qw     = (const float*)d_in[6];
  const float* kw     = (const float*)d_in[7];
  float* out = (float*)d_out;

  char* ws = (char*)d_ws;
  size_t off = 0;
  auto alloc = [&](size_t bytes) {
    char* p = ws + off;
    off += (bytes + 255) & ~(size_t)255;
    return (void*)p;
  };
  __bf16* Xb    = (__bf16*)alloc((size_t)BT * D_MODEL * 2);        // hidden bf16
  __bf16* WqkvT = (__bf16*)alloc((size_t)NQKV * D_MODEL * 2);      // [3072][2048]
  __bf16* WoT   = (__bf16*)alloc((size_t)D_MODEL * D_MODEL * 2);   // [2048][2048]
  float*  QKV   = (float*) alloc((size_t)BT * NQKV * 4);           // fp32 projections
  __bf16* Qb    = (__bf16*)alloc((size_t)BATCH * N_HEADS * SEQ * HD * 2);
  __bf16* Kb    = (__bf16*)alloc((size_t)BATCH * N_KV * SEQ * HD * 2);
  __bf16* VTb   = (__bf16*)alloc((size_t)BATCH * N_KV * HD * SEQ * 2);
  __bf16* Ab    = (__bf16*)alloc((size_t)BT * D_MODEL * 2);        // attention out bf16

  // 1) hidden -> bf16
  cvt_bf16_kernel<<<(BT * D_MODEL) / 1024, 256, 0, stream>>>(hidden, Xb, BT * D_MODEL);

  // 2) weight transposes into K-contiguous bf16
  transpose_cvt_kernel<<<dim3(D_MODEL / 32, D_MODEL / 32), dim3(32, 8), 0, stream>>>(Wq, WqkvT, D_MODEL, 0);
  transpose_cvt_kernel<<<dim3(512 / 32,    D_MODEL / 32), dim3(32, 8), 0, stream>>>(Wk, WqkvT, 512, 2048);
  transpose_cvt_kernel<<<dim3(512 / 32,    D_MODEL / 32), dim3(32, 8), 0, stream>>>(Wv, WqkvT, 512, 2560);
  transpose_cvt_kernel<<<dim3(D_MODEL / 32, D_MODEL / 32), dim3(32, 8), 0, stream>>>(Wo, WoT, D_MODEL, 0);

  // 3) fused QKV projection (WMMA)
  gemm_bf16_kernel<<<dim3(NQKV / 128, BT / 128), 128, 0, stream>>>(Xb, WqkvT, QKV, BT, NQKV, D_MODEL);

  // 4) RMSNorm + RoPE for q and k
  qk_norm_rope_kernel<<<dim3(BT, N_HEADS + N_KV), 128, 0, stream>>>(QKV, pos, qw, kw, Qb, Kb);

  // 5) V transpose to [d][t] bf16
  v_transpose_kernel<<<BATCH * N_KV * HD, 256, 0, stream>>>(QKV, VTb);

  // 6) flash attention (WMMA QK^T + PV)
  attn_kernel<<<dim3(SEQ / 64, N_HEADS, BATCH), 128, 0, stream>>>(Qb, Kb, VTb, Ab);

  // 7) output projection (WMMA) -> fp32 d_out
  gemm_bf16_kernel<<<dim3(D_MODEL / 128, BT / 128), 128, 0, stream>>>(Ab, WoT, out, BT, D_MODEL, D_MODEL);
}
